// EFAdLIF_74826920230997
// MI455X (gfx1250) — matrix operand order repaired and checked by reading, not compiled, not run on hardware
//
#include <hip/hip_runtime.h>
#include <hip/hip_bf16.h>
#include <stdint.h>

// ---------------- problem constants ----------------
#define T_STEPS 1000
#define BATCH   64
#define IN_F    512
#define OUT_F   512
#define THR_C   1.0f
#define Q_C     60.0f

// ---------------- CDNA5 WMMA types ----------------
typedef __attribute__((ext_vector_type(16))) __bf16          v16bf;
typedef __attribute__((ext_vector_type(16))) unsigned short  v16us;
typedef __attribute__((ext_vector_type(8)))  float           v8f;

union Frag {
    uint4  q[2];   // two 16B chunks
    v16us  us;
    v16bf  bf;
};

__device__ __forceinline__ unsigned short f2bfu(float f) {
    unsigned u = __float_as_uint(f);
    u += 0x7FFFu + ((u >> 16) & 1u);       // round-to-nearest-even
    return (unsigned short)(u >> 16);
}

// ---------------- kernel 0: bulk f32 -> bf16 (bandwidth-bound pre-pass) ----------------
// 8 elements per thread: 2x b128 load, 1x b128 store.
__global__ void cvt_bf16x8(const float* __restrict__ in,
                           unsigned short* __restrict__ out, int n8) {
    int i = blockIdx.x * blockDim.x + threadIdx.x;
    if (i >= n8) return;
    const float4 a = ((const float4*)in)[2 * i];
    const float4 b = ((const float4*)in)[2 * i + 1];
    union { uint4 q; unsigned short us[8]; } p;
    p.us[0] = f2bfu(a.x); p.us[1] = f2bfu(a.y); p.us[2] = f2bfu(a.z); p.us[3] = f2bfu(a.w);
    p.us[4] = f2bfu(b.x); p.us[5] = f2bfu(b.y); p.us[6] = f2bfu(b.z); p.us[7] = f2bfu(b.w);
    ((uint4*)out)[i] = p.q;
}

// ---------------- kernel 1: cur[t,b,o] = input[b,t,:]·W^T + bias ----------------
// All-bf16 inner loop: one wave per 16x32 tile (2 accumulators share the A frag).
// 6x global_load_b128 + 2x v_wmma_f32_16x16x32_bf16 per K-iteration, no cvt VALU.
__global__ void __launch_bounds__(512) ff_gemm(const unsigned short* __restrict__ inbf,
                                               const unsigned short* __restrict__ wgtbf,
                                               const float* __restrict__ bias,
                                               float* __restrict__ cur) {
    const int lane  = threadIdx.x & 31;
    const int wave  = threadIdx.x >> 5;
    const int mrow  = lane & 15;
    const int lh    = lane >> 4;                 // lane half (0/1)
    const int gtile = blockIdx.x * 16 + wave;    // 64000 16x32 tiles total
    const int rt    = gtile >> 4;                // row tile (4000)
    const int cp    = gtile & 15;                // column pair (16)
    const int r0    = rt * 16;
    const int n0    = cp * 32;
    const int n1    = n0 + 16;

    // A row for this lane: flat row r = t*BATCH + b  ->  input[b][t][:]
    const int r = r0 + mrow;
    const int b = r & (BATCH - 1);
    const int t = r >> 6;
    const unsigned short* aBase  = inbf  + ((size_t)b * T_STEPS + t) * IN_F;
    const unsigned short* b0Base = wgtbf + (size_t)(n0 + mrow) * IN_F;
    const unsigned short* b1Base = wgtbf + (size_t)(n1 + mrow) * IN_F;
    const float bv0 = bias[n0 + mrow];
    const float bv1 = bias[n1 + mrow];

    v8f acc0 = {}, acc1 = {};
    #pragma unroll
    for (int kb = 0; kb < IN_F / 32; ++kb) {
        const int k0 = kb * 32;
        Frag fa, f0, f1;
        // A: lane holds row mrow, K = 16 contiguous values (+16 if lh)
        const int klo = k0 + lh * 16;
        fa.q[0] = *(const uint4*)(aBase + klo);
        fa.q[1] = *(const uint4*)(aBase + klo + 8);
        // B: lane holds a column, K = 16 contiguous values (+16 if lh)
        const int kbb = k0 + lh * 16;
        f0.q[0] = *(const uint4*)(b0Base + kbb);
        f0.q[1] = *(const uint4*)(b0Base + kbb + 8);
        f1.q[0] = *(const uint4*)(b1Base + kbb);
        f1.q[1] = *(const uint4*)(b1Base + kbb + 8);
        acc0 = __builtin_amdgcn_wmma_f32_16x16x32_bf16(false, fa.bf, false, f0.bf,
                                                       (short)0, acc0, false, false);
        acc1 = __builtin_amdgcn_wmma_f32_16x16x32_bf16(false, fa.bf, false, f1.bf,
                                                       (short)0, acc1, false, false);
    }
    // C/D layout: lane holds col, VGPR j holds row j + 8*lh
    #pragma unroll
    for (int j = 0; j < 8; ++j) {
        const int rr = r0 + j + lh * 8;
        cur[(size_t)rr * OUT_F + n0 + mrow] = acc0[j] + bv0;
        cur[(size_t)rr * OUT_F + n1 + mrow] = acc1[j] + bv1;
    }
}

// ---------------- kernel 2: sequential LIF scan ----------------
// 4 workgroups (16 batch rows each) x 512 threads (16 waves).
// Wave handles 2 column tiles -> 2 interleaved WMMA chains per step.
#define ZSTRIDE 520   // 16B-aligned row stride with bank skew

__global__ void __launch_bounds__(512) lif_scan(const float* __restrict__ cur,
                                                const unsigned short* __restrict__ recbf,
                                                const float* __restrict__ Ap,
                                                const float* __restrict__ Bp,
                                                const float* __restrict__ DUp,
                                                const float* __restrict__ DWp,
                                                float* __restrict__ out) {
    __shared__ unsigned short zb[2][16 * ZSTRIDE];  // double-buffered z (bf16 A layout)

    const int lane = threadIdx.x & 31;
    const int wave = threadIdx.x >> 5;
    const int mrow = lane & 15;
    const int lh   = lane >> 4;
    const int b0   = blockIdx.x * 16;
    const int n0s[2] = { wave * 32, wave * 32 + 16 };

    float pa[2], pb[2], pdu[2], pdw[2];
    #pragma unroll
    for (int s = 0; s < 2; ++s) {
        const int n = n0s[s] + mrow;
        pa[s] = Ap[n]; pb[s] = Bp[n]; pdu[s] = DUp[n]; pdw[s] = DWp[n];
    }

    float u[2][8] = {}, w[2][8] = {}, zf[2][8] = {};

    for (int i = threadIdx.x; i < 16 * ZSTRIDE; i += 512) zb[0][i] = 0;
    __syncthreads();

    int p = 0;
    for (int t = 0; t < T_STEPS; ++t) {
        // feed-forward current for this step (independent of LDS -> issued early)
        const float* cb = cur + ((size_t)t * BATCH + b0) * OUT_F;
        float c0[8], c1[8];
        #pragma unroll
        for (int j = 0; j < 8; ++j) {
            const int m = j + lh * 8;
            c0[j] = cb[(size_t)m * OUT_F + n0s[0] + mrow];
            c1[j] = cb[(size_t)m * OUT_F + n0s[1] + mrow];
        }
        if (t + 1 < T_STEPS) {   // hide HBM latency on the serial path
            const float* cn = cb + (size_t)BATCH * OUT_F;
            #pragma unroll
            for (int j = 0; j < 8; ++j)
                __builtin_prefetch(cn + (size_t)(j + lh * 8) * OUT_F + n0s[0] + mrow, 0, 1);
        }

        // soma = z_{t-1} @ recurrent^T : two interleaved WMMA chains over K=512
        v8f acc0 = {}, acc1 = {};
        #pragma unroll
        for (int kb = 0; kb < OUT_F / 32; ++kb) {
            const int k0 = kb * 32;
            Frag fa, f0, f1;
            const int klo = k0 + lh * 16;
            fa.q[0] = *(const uint4*)&zb[p][mrow * ZSTRIDE + klo];
            fa.q[1] = *(const uint4*)&zb[p][mrow * ZSTRIDE + klo + 8];
            const int kbb = k0 + lh * 16;
            f0.q[0] = *(const uint4*)(recbf + (size_t)(n0s[0] + mrow) * OUT_F + kbb);
            f0.q[1] = *(const uint4*)(recbf + (size_t)(n0s[0] + mrow) * OUT_F + kbb + 8);
            f1.q[0] = *(const uint4*)(recbf + (size_t)(n0s[1] + mrow) * OUT_F + kbb);
            f1.q[1] = *(const uint4*)(recbf + (size_t)(n0s[1] + mrow) * OUT_F + kbb + 8);
            acc0 = __builtin_amdgcn_wmma_f32_16x16x32_bf16(false, fa.bf, false, f0.bf,
                                                           (short)0, acc0, false, false);
            acc1 = __builtin_amdgcn_wmma_f32_16x16x32_bf16(false, fa.bf, false, f1.bf,
                                                           (short)0, acc1, false, false);
        }

        // elementwise LIF update in the C/D register layout
        #pragma unroll
        for (int s = 0; s < 2; ++s) {
            v8f&  acc = s ? acc1 : acc0;
            float* cc = s ? c1   : c0;
            const int n = n0s[s] + mrow;
            #pragma unroll
            for (int j = 0; j < 8; ++j) {
                const float zp   = zf[s][j];
                const float ur   = u[s][j] * (1.0f - zp);          // detached reset
                const float soma = cc[j] + acc[j];
                const float du   = pdu[s];
                const float un   = du * ur + (1.0f - du) * (soma - w[s][j]);
                const float zn   = (un - THR_C > 0.0f) ? 1.0f : 0.0f;
                const float wn   = pdw[s] * w[s][j]
                                 + (1.0f - pdw[s]) * (pa[s] * ur + pb[s] * zp) * Q_C;
                u[s][j] = un; w[s][j] = wn; zf[s][j] = zn;
                const int m = j + lh * 8;
                zb[p ^ 1][m * ZSTRIDE + n] =
                    (zn > 0.0f) ? (unsigned short)0x3F80 : (unsigned short)0;
                out[((size_t)(b0 + m) * T_STEPS + t) * OUT_F + n] = zn;
            }
        }
        __syncthreads();   // single barrier per step thanks to double buffering
        p ^= 1;
    }
}

// ---------------- launch ----------------
extern "C" void kernel_launch(void* const* d_in, const int* in_sizes, int n_in,
                              void* d_out, int out_size, void* d_ws, size_t ws_size,
                              hipStream_t stream) {
    const float* input  = (const float*)d_in[0];
    const float* weight = (const float*)d_in[1];
    const float* bias   = (const float*)d_in[2];
    const float* rec    = (const float*)d_in[3];
    const float* a      = (const float*)d_in[4];
    const float* b      = (const float*)d_in[5];
    const float* du     = (const float*)d_in[6];
    const float* dw     = (const float*)d_in[7];
    float* out = (float*)d_out;

    // workspace layout:
    //   cur   : T*B*O f32      = 131,072,000 B
    //   recbf : O*O bf16       =     524,288 B
    //   inbf  : B*T*I bf16     =  65,536,000 B
    //   wgtbf : O*I bf16       =     524,288 B
    char* ws = (char*)d_ws;
    float*          curp  = (float*)ws;
    unsigned short* recbf = (unsigned short*)(ws + (size_t)T_STEPS * BATCH * OUT_F * 4);
    unsigned short* inbf  = (unsigned short*)((char*)recbf + (size_t)OUT_F * OUT_F * 2);
    unsigned short* wgtbf = (unsigned short*)((char*)inbf + (size_t)BATCH * T_STEPS * IN_F * 2);

    // pre-pass conversions (bandwidth-bound, ~200 MB total traffic ~ 10 us)
    {
        const int n8_in  = BATCH * T_STEPS * IN_F / 8;   // 4,096,000
        const int n8_mat = OUT_F * OUT_F / 8;            // 32,768
        cvt_bf16x8<<<(n8_in  + 255) / 256, 256, 0, stream>>>(input,  inbf,  n8_in);
        cvt_bf16x8<<<(n8_mat + 255) / 256, 256, 0, stream>>>(weight, wgtbf, n8_mat);
        cvt_bf16x8<<<(n8_mat + 255) / 256, 256, 0, stream>>>(rec,    recbf, n8_mat);
    }

    // 64000 16x32 tiles, 16 waves per block -> 4000 blocks
    ff_gemm<<<(T_STEPS * BATCH / 16) * (OUT_F / 32) / 16, 512, 0, stream>>>(
        inbf, wgtbf, bias, curp);

    lif_scan<<<BATCH / 16, 512, 0, stream>>>(curp, recbf, a, b, du, dw, out);
}
